// DrugEncoder2Real2Imag_31293131719205
// MI455X (gfx1250) — matrix operand order repaired and checked by reading, not compiled, not run on hardware
//
#include <hip/hip_runtime.h>
#include <stdint.h>

#define NND 100000
#define NED 1600000
#define NGR 2000
#define INDIM 15
#define HID 64
#define OUTD 128
#define AKT 4        // activation buffers: 4 k-tiles (128 cols) in packed layout

typedef __attribute__((ext_vector_type(16))) __bf16 v16bf;
typedef __attribute__((ext_vector_type(8)))  float  v8f;

__device__ __forceinline__ unsigned short f32_to_bf16_rne(float f) {
  unsigned int u = __float_as_uint(f);
  u += 0x7FFFu + ((u >> 16) & 1u);
  return (unsigned short)(u >> 16);
}
__device__ __forceinline__ float bf16_to_f32(unsigned short h) {
  return __uint_as_float(((unsigned int)h) << 16);
}

// A-fragment packed layout (CDNA5 ISA 7.12.2, 16-bit A 16x32):
//   lanes 0-15 : K 0..7 (h0-7), K 16..23 (h8-15);  lanes 16-31: K 8..15, K 24..31
// element (node,f) -> half index within buffer of `akt` k-tiles per 16-row tile
__device__ __forceinline__ int packA_idx(int node, int f, int akt) {
  int mt = node >> 4, row = node & 15;
  int kt = f >> 5, kin = f & 31;
  int lane = row + (((kin >> 3) & 1) << 4);
  int h    = (kin & 7) + ((kin >> 4) << 3);
  return ((mt * akt + kt) * 32 + lane) * 16 + h;
}

// ---------------- packing kernels ----------------
// input features -> A-fragment layout, K padded 15->32, 1 k-tile per row tile
__global__ void pack_X(const float* __restrict__ x, unsigned short* __restrict__ out, int n) {
  int i = blockIdx.x * blockDim.x + threadIdx.x;        // n*32 threads
  if (i >= n * 32) return;
  int h = i & 15, lane = (i >> 4) & 31, mt = i >> 9;
  int row = lane & 15;
  int kin = (h & 7) + ((h >> 3) << 4) + ((lane >> 4) << 3);
  int node = mt * 16 + row;
  out[i] = (kin < INDIM) ? f32_to_bf16_rne(x[node * INDIM + kin]) : (unsigned short)0;
}

// weights f32[K x dout] -> B-fragment layout (32x16 bf16 B: lane=col;
// lanes 0-15: K 0..7, 16..23; lanes 16-31: K 8..15, 24..31), zero-padded past K
__global__ void pack_B(const float* __restrict__ W, unsigned short* __restrict__ out,
                       int K, int dout, int ktiles) {
  int i = blockIdx.x * blockDim.x + threadIdx.x;        // ktiles*(dout/16)*512 threads
  int ctiles = dout >> 4;
  if (i >= ktiles * ctiles * 512) return;
  int h = i & 15, lane = (i >> 4) & 31, ntk = i >> 9;
  int kt = ntk / ctiles, nt = ntk - kt * ctiles;
  int col = nt * 16 + (lane & 15);
  int kb = kt * 32 + (h & 7) + ((h >> 3) << 4) + ((lane >> 4) << 3);
  out[i] = (kb < K) ? f32_to_bf16_rne(W[kb * dout + col]) : (unsigned short)0;
}

// ---------------- utility ----------------
__global__ void zero_f32(float* __restrict__ p, int n) {
  int i = blockIdx.x * blockDim.x + threadIdx.x;
  if (i < n) p[i] = 0.0f;
}
__global__ void fill_f32(float* __restrict__ p, float v, int n) {
  int i = blockIdx.x * blockDim.x + threadIdx.x;
  if (i < n) p[i] = v;
}

// ---------------- degree / norm ----------------
__global__ void deg_accum(const int* __restrict__ dst, float* __restrict__ deg, int ne) {
  int i = blockIdx.x * blockDim.x + threadIdx.x;
  if (i < ne) atomicAdd(&deg[dst[i]], 1.0f);
}
__global__ void make_dinv(float* __restrict__ deg, int n) {
  int i = blockIdx.x * blockDim.x + threadIdx.x;
  if (i < n) deg[i] = rsqrtf(deg[i]);   // deg >= 1 (self loop)
}
__global__ void make_norm(const int* __restrict__ src, const int* __restrict__ dst,
                          const float* __restrict__ dinv, float* __restrict__ norm, int ne) {
  int i = blockIdx.x * blockDim.x + threadIdx.x;
  if (i < ne) norm[i] = dinv[src[i]] * dinv[dst[i]];
}

// ---------------- WMMA GEMM on fragment-packed operands ----------------
// One wave owns a 16-row tile, loads its A fragments once (2x b128 per k-tile),
// then sweeps all column tiles: per (kt,nt) two b128 B loads + one
// v_wmma_f32_16x16x32_bf16.  Fully unrolled, no guards: EXEC all-1 throughout.
template <int KTILES, int CTILES, bool PACK_OUT>
__global__ __launch_bounds__(256) void gemm_wmma(
    const uint4* __restrict__ Apk, int aktiles,
    const uint4* __restrict__ Bpk,
    const float* __restrict__ bias,
    float* __restrict__ outF,
    unsigned short* __restrict__ outPk,
    int rows) {
  const int lane = threadIdx.x & 31;
  const int mt = blockIdx.x * (blockDim.x >> 5) + (threadIdx.x >> 5);
  if (mt * 16 >= rows) return;          // wave-uniform exit

  union Frag { uint4 q[2]; v16bf v; };
  Frag a[KTILES];
#pragma unroll
  for (int kt = 0; kt < KTILES; ++kt) {
    size_t ai = ((size_t)(mt * aktiles + kt) * 32 + lane) * 2;
    a[kt].q[0] = Apk[ai];
    a[kt].q[1] = Apk[ai + 1];
  }

  const int mbase = mt * 16 + ((lane < 16) ? 0 : 8);
#pragma unroll
  for (int nt = 0; nt < CTILES; ++nt) {
    v8f acc = {};
#pragma unroll
    for (int kt = 0; kt < KTILES; ++kt) {
      Frag b;
      size_t bi = ((size_t)(kt * CTILES + nt) * 32 + lane) * 2;
      b.q[0] = Bpk[bi];
      b.q[1] = Bpk[bi + 1];
      acc = __builtin_amdgcn_wmma_f32_16x16x32_bf16(false, a[kt].v, false, b.v,
                                                    (short)0, acc, false, false);
    }
    const int col = nt * 16 + (lane & 15);
    const float bb = bias ? bias[col] : 0.0f;
#pragma unroll
    for (int r = 0; r < 8; ++r) {
      float v = acc[r] + bb;
      int node = mbase + r;
      if constexpr (PACK_OUT) {
        outPk[packA_idx(node, col, AKT)] = f32_to_bf16_rne(v);
      } else {
        outF[(size_t)node * (CTILES * 16) + col] = v;
      }
    }
  }
}

// ---------------- edge scatter-add: agg[dst] += norm * h[src] ----------------
__global__ void aggregate_edges(const int* __restrict__ src, const int* __restrict__ dst,
                                const float* __restrict__ norm, const float* __restrict__ h,
                                float* __restrict__ agg, int ne, int dout) {
  const int lane = threadIdx.x & 31;
  const int wid  = blockIdx.x * (blockDim.x >> 5) + (threadIdx.x >> 5);
  const int nw   = gridDim.x * (blockDim.x >> 5);
  for (int e = wid; e < ne; e += nw) {
    __builtin_prefetch(src + e + 4 * nw, 0, 1);   // global_prefetch_b8
    const int s = src[e], d = dst[e];
    const float w = norm[e];
    const float* hs = h   + (size_t)s * dout;
    float*       ad = agg + (size_t)d * dout;
    for (int f = lane; f < dout; f += 32)
      atomicAdd(&ad[f], w * hs[f]);
  }
}

// -------- fused bias + self-loop + BN + ReLU -> packed bf16 activation --------
__global__ void bn_relu_store(const float* __restrict__ agg, const float* __restrict__ h,
                              const float* __restrict__ dinv, const float* __restrict__ bias,
                              const float* __restrict__ gamma, const float* __restrict__ beta,
                              const float* __restrict__ mean, const float* __restrict__ var,
                              unsigned short* __restrict__ xout, int n, int dout, int dshift) {
  int i = blockIdx.x * blockDim.x + threadIdx.x;
  if (i >= n * dout) return;
  int node = i >> dshift, f = i & (dout - 1);
  float di = dinv[node];
  float v = agg[i] + di * di * h[i] + bias[f];
  float y = (v - mean[f]) * rsqrtf(var[f] + 1e-5f) * gamma[f] + beta[f];
  xout[packA_idx(node, f, AKT)] = f32_to_bf16_rne(fmaxf(y, 0.0f));
}

// ---------------- mean pool ----------------
__global__ void count_graphs(const int* __restrict__ batch, float* __restrict__ counts, int n) {
  int i = blockIdx.x * blockDim.x + threadIdx.x;
  if (i < n) atomicAdd(&counts[batch[i]], 1.0f);
}
__global__ void pool_accum(const unsigned short* __restrict__ x0, const unsigned short* __restrict__ x1,
                           const unsigned short* __restrict__ x2, const unsigned short* __restrict__ x3,
                           const int* __restrict__ batch, float* __restrict__ out, int n) {
  int i = blockIdx.x * blockDim.x + threadIdx.x;        // n * OUTD threads
  if (i >= n * OUTD) return;
  int node = i >> 7, f = i & (OUTD - 1);
  int g = batch[node];
  int src = packA_idx(node, f, AKT);
  size_t base = (size_t)g * OUTD + f;
  const size_t stride = (size_t)NGR * OUTD;
  atomicAdd(&out[base],              bf16_to_f32(x0[src]));
  atomicAdd(&out[base + stride],     bf16_to_f32(x1[src]));
  atomicAdd(&out[base + 2 * stride], bf16_to_f32(x2[src]));
  atomicAdd(&out[base + 3 * stride], bf16_to_f32(x3[src]));
}
__global__ void pool_div(float* __restrict__ out, const float* __restrict__ counts) {
  int i = blockIdx.x * blockDim.x + threadIdx.x;
  if (i >= 4 * NGR * OUTD) return;
  int g = (i / OUTD) % NGR;
  out[i] /= fmaxf(counts[g], 1.0f);
}

// ---------------- host launcher ----------------
static inline int gdiv(int a, int b) { return (a + b - 1) / b; }

extern "C" void kernel_launch(void* const* d_in, const int* in_sizes, int n_in,
                              void* d_out, int out_size, void* d_ws, size_t ws_size,
                              hipStream_t stream) {
  (void)in_sizes; (void)n_in; (void)out_size; (void)ws_size;
  // jax pytree flatten order (dict keys sorted):
  // 0:x 1:edge_index 2:batch 3:Wp1 4:Wp2 5:bp1 6:bp2
  // layer l (base=7+20l): W1,W2,b1,b2, bn_i1{beta,gamma,mean,var}, bn_i2, bn_r1, bn_r2
  const float* x     = (const float*)d_in[0];
  const int*   ei    = (const int*)d_in[1];
  const int*   batch = (const int*)d_in[2];
  const float* Wp1   = (const float*)d_in[3];
  const float* Wp2   = (const float*)d_in[4];
  const float* bp1   = (const float*)d_in[5];
  const float* bp2   = (const float*)d_in[6];
  const int* srcE = ei;
  const int* dstE = ei + NED;

  char* ws = (char*)d_ws;
  size_t off = 0;
  auto alloc = [&](size_t bytes) -> char* {
    char* p = ws + off;
    off = (off + bytes + 255) & ~(size_t)255;
    return p;
  };
  unsigned short* xbf = (unsigned short*)alloc((size_t)NND * 32 * 2);     // packed, 1 k-tile
  unsigned short* xb[4];
  for (int s = 0; s < 4; ++s) xb[s] = (unsigned short*)alloc((size_t)NND * 128 * 2); // packed, 4 k-tiles
  float* hbuf   = (float*)alloc((size_t)NND * OUTD * 4);
  float* agg    = (float*)alloc((size_t)NND * OUTD * 4);
  float* dinv   = (float*)alloc((size_t)NND * 4);
  float* norm   = (float*)alloc((size_t)NED * 4);
  float* counts = (float*)alloc((size_t)NGR * 4);
  unsigned short* Wp1b = (unsigned short*)alloc((size_t)1 * 4 * 512 * 2);
  unsigned short* Wp2b = (unsigned short*)alloc((size_t)1 * 4 * 512 * 2);
  unsigned short* W1b[3]; unsigned short* W2b[3];
  for (int l = 0; l < 3; ++l) {
    int ct = (l == 2) ? 8 : 4;
    W1b[l] = (unsigned short*)alloc((size_t)2 * ct * 512 * 2);
    W2b[l] = (unsigned short*)alloc((size_t)2 * ct * 512 * 2);
  }

  const int T = 256;

  // --- pack inputs / weights into WMMA fragment layouts
  pack_X<<<gdiv(NND * 32, T), T, 0, stream>>>(x, xbf, NND);
  pack_B<<<gdiv(1 * 4 * 512, T), T, 0, stream>>>(Wp1, Wp1b, INDIM, HID, 1);
  pack_B<<<gdiv(1 * 4 * 512, T), T, 0, stream>>>(Wp2, Wp2b, INDIM, HID, 1);
  for (int l = 0; l < 3; ++l) {
    int dl = (l == 2) ? OUTD : HID;
    int ct = dl >> 4;
    int base = 7 + 20 * l;
    pack_B<<<gdiv(2 * ct * 512, T), T, 0, stream>>>((const float*)d_in[base + 0], W1b[l], HID, dl, 2);
    pack_B<<<gdiv(2 * ct * 512, T), T, 0, stream>>>((const float*)d_in[base + 1], W2b[l], HID, dl, 2);
  }

  // --- degrees (incl. self loop) -> dinv -> per-edge norm
  fill_f32<<<gdiv(NND, T), T, 0, stream>>>(dinv, 1.0f, NND);
  deg_accum<<<gdiv(NED, T), T, 0, stream>>>(dstE, dinv, NED);
  make_dinv<<<gdiv(NND, T), T, 0, stream>>>(dinv, NND);
  make_norm<<<gdiv(NED, T), T, 0, stream>>>(srcE, dstE, dinv, norm, NED);

  const int gblocks = gdiv((NND / 16) * 32, T);   // one wave per 16-row tile

  // --- initial projections (bf16 WMMA, packed bf16 out); imag streams = copies
  gemm_wmma<1, 4, true><<<gblocks, T, 0, stream>>>(
      (const uint4*)xbf, 1, (const uint4*)Wp1b, bp1, nullptr, xb[0], NND);
  gemm_wmma<1, 4, true><<<gblocks, T, 0, stream>>>(
      (const uint4*)xbf, 1, (const uint4*)Wp2b, bp2, nullptr, xb[1], NND);
  hipMemcpyAsync(xb[2], xb[0], (size_t)NND * 128 * 2, hipMemcpyDeviceToDevice, stream);
  hipMemcpyAsync(xb[3], xb[1], (size_t)NND * 128 * 2, hipMemcpyDeviceToDevice, stream);

  // --- GCN layers: streams 0=r1(W1,bn_r1) 1=r2(W2,bn_r2) 2=i1(W1,bn_i1) 3=i2(W2,bn_i2)
  static const int bnoff[4] = {12, 16, 4, 8};
  for (int l = 0; l < 3; ++l) {
    const int dout = (l == 2) ? OUTD : HID;
    const int base = 7 + 20 * l;
    const float* b1 = (const float*)d_in[base + 2];
    const float* b2 = (const float*)d_in[base + 3];
    for (int s = 0; s < 4; ++s) {
      const unsigned short* Wb = (s & 1) ? W2b[l] : W1b[l];
      const float* bb = (s & 1) ? b2 : b1;
      const float* bn_beta  = (const float*)d_in[base + bnoff[s] + 0];
      const float* bn_gamma = (const float*)d_in[base + bnoff[s] + 1];
      const float* bn_mean  = (const float*)d_in[base + bnoff[s] + 2];
      const float* bn_var   = (const float*)d_in[base + bnoff[s] + 3];

      if (dout == HID)
        gemm_wmma<2, 4, false><<<gblocks, T, 0, stream>>>(
            (const uint4*)xb[s], AKT, (const uint4*)Wb, nullptr, hbuf, nullptr, NND);
      else
        gemm_wmma<2, 8, false><<<gblocks, T, 0, stream>>>(
            (const uint4*)xb[s], AKT, (const uint4*)Wb, nullptr, hbuf, nullptr, NND);

      zero_f32<<<gdiv(NND * dout, T), T, 0, stream>>>(agg, NND * dout);
      aggregate_edges<<<4096, T, 0, stream>>>(srcE, dstE, norm, hbuf, agg, NED, dout);
      bn_relu_store<<<gdiv(NND * dout, T), T, 0, stream>>>(
          agg, hbuf, dinv, bb, bn_gamma, bn_beta, bn_mean, bn_var,
          xb[s], NND, dout, (dout == 64) ? 6 : 7);
    }
  }

  // --- mean pool per graph, outputs concatenated r1,r2,i1,i2
  float* out = (float*)d_out;
  zero_f32<<<gdiv(4 * NGR * OUTD, T), T, 0, stream>>>(out, 4 * NGR * OUTD);
  zero_f32<<<gdiv(NGR, T), T, 0, stream>>>(counts, NGR);
  count_graphs<<<gdiv(NND, T), T, 0, stream>>>(batch, counts, NND);
  pool_accum<<<gdiv(NND * OUTD, T), T, 0, stream>>>(xb[0], xb[1], xb[2], xb[3], batch, out, NND);
  pool_div<<<gdiv(4 * NGR * OUTD, T), T, 0, stream>>>(out, counts);
}
